// UltraEfficientSparseFFN_44487271252147
// MI455X (gfx1250) — matrix operand order repaired and checked by reading, not compiled, not run on hardware
//
#include <hip/hip_runtime.h>
#include <hip/hip_bf16.h>

// ---------------- problem constants ----------------
#define BB 8
#define TT 4096
#define DD 1024
#define ROWS (BB * TT)          // 32768
#define NBINS 513               // rfft bins for D=1024
#define KFREQ 128
#define POLY_KEEP 512
#define MICRO_KEEP 256
#define EPSV 1e-05f

typedef __attribute__((ext_vector_type(16))) _Float16 v16h;
typedef __attribute__((ext_vector_type(8)))  float    v8f;

// WMMA-friendly K permutation within a 32-wide chunk: swap bits 3 and 4.
// Chunk memory order = K {0..7, 16..23, 8..15, 24..31}, so an A fragment is
// one contiguous 32B load per lane (lanes 0-15 -> first 16, lanes 16-31 -> last 16).
__device__ __forceinline__ int kperm(int d) {
    int p = d & 31;
    return (d - p) + (p & 7) + ((p & 8) << 1) + ((p & 16) >> 1);
}

// ---------------- block reduction helper ----------------
__device__ __forceinline__ float block_sum(float v, float* red, int tid) {
    red[tid] = v;
    __syncthreads();
    for (int off = 128; off > 0; off >>= 1) {
        if (tid < off) red[tid] += red[tid + off];
        __syncthreads();
    }
    float r = red[0];
    __syncthreads();
    return r;
}

// ---------------- kernel 1: channel masks (fixed per dispatch) ----------------
__global__ __launch_bounds__(256) void topk_mask_kernel(
    const float* __restrict__ poly_imp, const float* __restrict__ micro_imp,
    unsigned char* __restrict__ poly_mask, unsigned char* __restrict__ micro_mask) {
    __shared__ float a[DD];
    int tid = threadIdx.x;

    for (int i = tid; i < DD; i += 256) a[i] = poly_imp[i];
    __syncthreads();
    for (int i = tid; i < DD; i += 256) {
        float vi = a[i];
        int cnt = 0;
        for (int j = 0; j < DD; j++) {
            float vj = a[j];
            cnt += (vj > vi) || (vj == vi && j < i);
        }
        poly_mask[i] = (cnt < POLY_KEEP) ? 1 : 0;
    }
    __syncthreads();
    for (int i = tid; i < DD; i += 256) a[i] = micro_imp[i];
    __syncthreads();
    for (int i = tid; i < DD; i += 256) {
        float vi = a[i];
        int cnt = 0;
        for (int j = 0; j < DD; j++) {
            float vj = a[j];
            cnt += (vj > vi) || (vj == vi && j < i);
        }
        micro_mask[i] = (cnt < MICRO_KEEP) ? 1 : 0;
    }
}

// ---------------- kernel 2: split proj_w into f16 hi/lo (keep [e][d] layout) ----------------
__global__ __launch_bounds__(256) void wsplit_kernel(
    const float* __restrict__ W, _Float16* __restrict__ Whi, _Float16* __restrict__ Wlo) {
    size_t i = (size_t)blockIdx.x * 256 + threadIdx.x;
    float w = W[i];
    _Float16 hi = (_Float16)w;
    Whi[i] = hi;
    Wlo[i] = (_Float16)(w - (float)hi);
}

// ---------------- kernel 3: per-row spectral pipeline ----------------
// LN -> FFT(1024, DIT, bit-rev in) -> bitonic top-128 gate (gains[rank]) ->
// hermitian fill -> inverse FFT (DIF, bit-rev out) -> +bias -> poly/micro masks
// -> LN -> pre-split f16 hi/lo A-operand in natural K layout.
__global__ __launch_bounds__(256) void spectral_kernel(
    const float* __restrict__ x,
    const float* __restrict__ ln_in_w, const float* __restrict__ ln_in_b,
    const float* __restrict__ gains,   const float* __restrict__ spec_bias,
    const float* __restrict__ coeffs,
    const float* __restrict__ micro_w, const float* __restrict__ micro_b,
    const float* __restrict__ ln_out_w, const float* __restrict__ ln_out_b,
    const unsigned char* __restrict__ poly_mask,
    const unsigned char* __restrict__ micro_mask,
    _Float16* __restrict__ Ahi, _Float16* __restrict__ Alo) {
    __shared__ float re[DD];
    __shared__ float im[DD];
    __shared__ unsigned long long key[DD];
    __shared__ float gfac[NBINS];
    __shared__ float red[256];

    const int tid = threadIdx.x;
    const size_t row = blockIdx.x;
    const float* xr = x + row * DD;

    // ---- input LayerNorm ----
    float v[4];
    float ps = 0.f;
    for (int i = 0; i < 4; i++) { v[i] = xr[tid + i * 256]; ps += v[i]; }
    float mean = block_sum(ps, red, tid) * (1.0f / DD);
    float ps2 = 0.f;
    for (int i = 0; i < 4; i++) { float dmu = v[i] - mean; ps2 += dmu * dmu; }
    float var = block_sum(ps2, red, tid) * (1.0f / DD);
    float rs = rsqrtf(var + EPSV);
    // store bit-reversed for DIT FFT
    for (int i = 0; i < 4; i++) {
        int d = tid + i * 256;
        float h = (v[i] - mean) * rs * ln_in_w[d] + ln_in_b[d];
        int rb = (int)(__brev((unsigned)d) >> 22);
        re[rb] = h;
        im[rb] = 0.f;
    }
    __syncthreads();

    // ---- forward FFT (radix-2 DIT), natural-order output ----
    for (int lh = 0; lh < 10; lh++) {
        int half = 1 << lh;
        for (int j = tid; j < (DD / 2); j += 256) {
            int pos  = j & (half - 1);
            int base = ((j >> lh) << (lh + 1)) + pos;
            float ang = -6.28318530717958647692f * (float)pos / (float)(2 << lh);
            float sn, cs;
            __sincosf(ang, &sn, &cs);
            float ar = re[base],        ai = im[base];
            float br = re[base + half], bi = im[base + half];
            float tr = br * cs - bi * sn;
            float ti = br * sn + bi * cs;
            re[base]        = ar + tr;  im[base]        = ai + ti;
            re[base + half] = ar - tr;  im[base + half] = ai - ti;
        }
        __syncthreads();
    }

    // ---- build sort keys: (|X|^2 bits << 32) | (1023 - bin); squared magnitude
    // is rank-monotonic and ties break toward lower bin, matching lax.top_k ----
    for (int f = tid; f < DD; f += 256) {
        if (f < NBINS) {
            float m2 = re[f] * re[f] + im[f] * im[f];
            key[f] = ((unsigned long long)__float_as_uint(m2) << 32) |
                     (unsigned long long)(unsigned)(1023 - f);
        } else {
            key[f] = (unsigned long long)(unsigned)(1023 - f);
        }
        if (f < NBINS) gfac[f] = 0.f;
    }

    // ---- bitonic sort, descending ----
    for (int k = 2; k <= DD; k <<= 1) {
        for (int j = k >> 1; j > 0; j >>= 1) {
            __syncthreads();
            for (int t4 = 0; t4 < 4; t4++) {
                int i = tid + t4 * 256;
                int l = i ^ j;
                if (l > i) {
                    unsigned long long ki = key[i], kl = key[l];
                    bool asc = (i & k) != 0;           // ascending segment
                    bool sw  = asc ? (ki > kl) : (ki < kl);
                    if (sw) { key[i] = kl; key[l] = ki; }
                }
            }
        }
    }
    __syncthreads();

    // ---- top-128: scatter gains[rank] into gate factors ----
    if (tid < KFREQ) {
        int f = 1023 - (int)(unsigned)(key[tid] & 0xFFFFFFFFull);
        gfac[f] = gains[tid];
    }
    __syncthreads();

    // ---- apply gate ----
    for (int f = tid; f < NBINS; f += 256) {
        float g = gfac[f];
        re[f] *= g;
        im[f] *= g;
    }
    __syncthreads();

    // ---- hermitian fill for full inverse transform ----
    for (int f = NBINS + tid; f < DD; f += 256) {
        re[f] =  re[DD - f];
        im[f] = -im[DD - f];
    }
    __syncthreads();

    // ---- inverse FFT (radix-2 DIF), bit-reversed output ----
    for (int lh = 9; lh >= 0; lh--) {
        int half = 1 << lh;
        for (int j = tid; j < (DD / 2); j += 256) {
            int pos  = j & (half - 1);
            int base = ((j >> lh) << (lh + 1)) + pos;
            float ang = 6.28318530717958647692f * (float)pos / (float)(2 << lh);
            float sn, cs;
            __sincosf(ang, &sn, &cs);
            float ar = re[base],        ai = im[base];
            float br = re[base + half], bi = im[base + half];
            float dr = ar - br, di = ai - bi;
            re[base]        = ar + br;
            im[base]        = ai + bi;
            re[base + half] = dr * cs - di * sn;
            im[base + half] = dr * sn + di * cs;
        }
        __syncthreads();
    }

    // ---- bias + poly + micro + output LayerNorm ----
    const float c0 = coeffs[0], c1 = coeffs[1], c2 = coeffs[2];
    const float mw0 = micro_w[0], mw1 = micro_w[1];
    const float mb0 = micro_b[0], mb1 = micro_b[1];

    float o[4];
    float ps3 = 0.f;
    for (int i = 0; i < 4; i++) {
        int d  = tid + i * 256;
        int rb = (int)(__brev((unsigned)d) >> 22);
        float h2 = re[rb] * (1.0f / DD) + spec_bias[d];
        if (poly_mask[d]) {
            float xv = h2;
            h2 = xv * (c0 + xv * (c1 + xv * c2));   // c0*x + c1*x^2 + c2*x^3
        }
        if (micro_mask[d]) {
            float xt = h2;
            float z0 = xt * mw0 + mb0;
            xt = z0 / (1.f + __expf(-z0));
            float z1 = xt * mw1 + mb1;
            xt = z1 / (1.f + __expf(-z1));
            h2 = xt;
        }
        o[i] = h2;
        ps3 += h2;
    }
    float mean2 = block_sum(ps3, red, tid) * (1.0f / DD);
    float ps4 = 0.f;
    for (int i = 0; i < 4; i++) { float dmu = o[i] - mean2; ps4 += dmu * dmu; }
    float var2 = block_sum(ps4, red, tid) * (1.0f / DD);
    float rs2 = rsqrtf(var2 + EPSV);

    // ---- pre-split f16 hi/lo, stored in natural K order ----
    _Float16* Ah = Ahi + row * DD;
    _Float16* Al = Alo + row * DD;
    for (int i = 0; i < 4; i++) {
        int d = tid + i * 256;
        float val = (o[i] - mean2) * rs2 * ln_out_w[d] + ln_out_b[d];
        _Float16 hi = (_Float16)val;
        int pos = d;
        Ah[pos] = hi;
        Al[pos] = (_Float16)(val - (float)hi);
    }
}

// ---------------- kernel 4: split-f16 WMMA GEMM + gated residual ----------------
// out[r,e] = x[r,e] + gate * ( sum_d H[r,d]*W[e,d] + proj_b[e] )
// H ~ Ahi+Alo (f16, pre-split, natural K order), W ~ Whi+Wlo (f16):
// acc = hi*whi + hi*wlo + lo*whi  (fp32-class accuracy, 3 f16 WMMAs)
// Block: 256 threads = 8 wave32s in a 4(M) x 2(N) layout.
// Wave tile = 32x64 (2x4 WMMA subtiles) -> block tile 128x128.
// 12 fragment loads feed 24 WMMAs per K-step per wave.
__global__ __launch_bounds__(256) void gemm_kernel(
    const _Float16* __restrict__ Ahi,   // [ROWS, DD] f16 (natural K order)
    const _Float16* __restrict__ Alo,
    const _Float16* __restrict__ Whi,   // [DD, DD]  f16, [e][d]
    const _Float16* __restrict__ Wlo,
    const float*    __restrict__ x,     // [ROWS, DD]
    const float*    __restrict__ proj_b,
    const float*    __restrict__ gate,
    float*          __restrict__ out) {
    const int lane = threadIdx.x & 31;
    const int wave = threadIdx.x >> 5;
    const int wm = wave & 3;         // 0..3  (M)
    const int wn = wave >> 2;        // 0..1  (N)
    const int mwave = blockIdx.y * 128 + wm * 32;
    const int nwave = blockIdx.x * 128 + wn * 64;

    const int l15  = lane & 15;
    const int grp  = (lane >> 4) & 1;   // lane group
    const int kof  = grp * 16;          // contiguous 16 halfs per lane (A and B)

    v8f c[2][4] = {};
    const float gt = gate[0];

    for (int k0 = 0; k0 < DD; k0 += 32) {
        v16h ahi[2], alo[2], bhi[4], blo[4];

        // A fragments: single 32B contiguous load per lane per matrix
        for (int t = 0; t < 2; t++) {
            const int m = mwave + t * 16 + l15;
            const size_t aoff = (size_t)m * DD + k0 + kof;
            ahi[t] = *(const v16h*)(Ahi + aoff);
            alo[t] = *(const v16h*)(Alo + aoff);
            __builtin_prefetch((const void*)(Ahi + aoff + 32), 0, 0);
            __builtin_prefetch((const void*)(Alo + aoff + 32), 0, 0);
        }
        // B fragments: contiguous 32B per lane from W rows (contraction over d)
        for (int u = 0; u < 4; u++) {
            const int n = nwave + u * 16 + l15;
            const size_t boff = (size_t)n * DD + k0 + kof;
            bhi[u] = *(const v16h*)(Whi + boff);
            blo[u] = *(const v16h*)(Wlo + boff);
        }

        for (int t = 0; t < 2; t++) {
            for (int u = 0; u < 4; u++) {
                c[t][u] = __builtin_amdgcn_wmma_f32_16x16x32_f16(
                    false, ahi[t], false, bhi[u], (short)0, c[t][u], false, false);
                c[t][u] = __builtin_amdgcn_wmma_f32_16x16x32_f16(
                    false, ahi[t], false, blo[u], (short)0, c[t][u], false, false);
                c[t][u] = __builtin_amdgcn_wmma_f32_16x16x32_f16(
                    false, alo[t], false, bhi[u], (short)0, c[t][u], false, false);
            }
        }
    }

    // Epilogue per C/D layout: lanes 0-15 N=lane (M base 0), lanes 16-31 N=lane-16 (M base 8)
    const int moff = grp * 8;
    for (int t = 0; t < 2; t++) {
        for (int u = 0; u < 4; u++) {
            const int n = nwave + u * 16 + l15;
            const float pb = proj_b[n];
            for (int r = 0; r < 8; r++) {
                const int m = mwave + t * 16 + moff + r;
                const size_t off = (size_t)m * DD + n;
                out[off] = x[off] + gt * (c[t][u][r] + pb);
            }
        }
    }
}

// ---------------- launcher ----------------
extern "C" void kernel_launch(void* const* d_in, const int* in_sizes, int n_in,
                              void* d_out, int out_size, void* d_ws, size_t ws_size,
                              hipStream_t stream) {
    const float* x         = (const float*)d_in[0];
    const float* ln_in_w   = (const float*)d_in[1];
    const float* ln_in_b   = (const float*)d_in[2];
    const float* gains     = (const float*)d_in[3];
    const float* spec_bias = (const float*)d_in[4];
    const float* coeffs    = (const float*)d_in[5];
    const float* poly_imp  = (const float*)d_in[6];
    const float* micro_imp = (const float*)d_in[7];
    const float* micro_w   = (const float*)d_in[8];
    const float* micro_b   = (const float*)d_in[9];
    const float* proj_w    = (const float*)d_in[10];
    const float* proj_b    = (const float*)d_in[11];
    const float* ln_out_w  = (const float*)d_in[12];
    const float* ln_out_b  = (const float*)d_in[13];
    const float* gate      = (const float*)d_in[14];
    float* out = (float*)d_out;

    // workspace layout
    _Float16* Ahi = (_Float16*)d_ws;                                  // ROWS*DD f16
    _Float16* Alo = Ahi + (size_t)ROWS * DD;                          // ROWS*DD f16
    _Float16* Whi = Alo + (size_t)ROWS * DD;                          // DD*DD f16
    _Float16* Wlo = Whi + (size_t)DD * DD;                            // DD*DD f16
    unsigned char* pmask = (unsigned char*)(Wlo + (size_t)DD * DD);   // DD bytes
    unsigned char* mmask = pmask + DD;                                // DD bytes

    topk_mask_kernel<<<1, 256, 0, stream>>>(poly_imp, micro_imp, pmask, mmask);
    wsplit_kernel<<<(DD * DD) / 256, 256, 0, stream>>>(proj_w, Whi, Wlo);
    spectral_kernel<<<ROWS, 256, 0, stream>>>(
        x, ln_in_w, ln_in_b, gains, spec_bias, coeffs,
        micro_w, micro_b, ln_out_w, ln_out_b, pmask, mmask, Ahi, Alo);
    dim3 grid(DD / 128, ROWS / 128);
    gemm_kernel<<<grid, 256, 0, stream>>>(Ahi, Alo, Whi, Wlo, x, proj_b, gate, out);
}
